// SetTransformerConditionEncoder_72911364817392
// MI455X (gfx1250) — matrix operand order repaired and checked by reading, not compiled, not run on hardware
//
#include <hip/hip_runtime.h>

typedef unsigned short ushort_t;
typedef __attribute__((ext_vector_type(16))) __bf16 v16bf;
typedef __attribute__((ext_vector_type(8)))  float  v8f;
typedef __attribute__((ext_vector_type(4)))  unsigned u32x4;
typedef __attribute__((ext_vector_type(8)))  int      i32x8;
typedef __attribute__((ext_vector_type(4)))  int      i32x4;

#define DEV __device__ __forceinline__

DEV float bf2f(ushort_t h) { return __uint_as_float(((unsigned)h) << 16); }
DEV ushort_t f2bf(float f) {
    unsigned u = __float_as_uint(f);
    u += 0x7FFFu + ((u >> 16) & 1u);          // round-to-nearest-even
    return (ushort_t)(u >> 16);
}
// CDNA5 WMMA bf16 fragment K-index maps (ISA 7.12.2)
DEV int kA(int j, int half) { return ((j >> 2) << 4) + half * 8 + ((j & 3) << 1); } // A 16x32
DEV int kB(int j, int half) { return ((j >> 2) << 4) + half * 8 + ((j & 3) << 1); } // B 32x16

union FragB16 { v16bf v; unsigned u[8]; };

// ---------------------------------------------------------------------------
// TDM: DMA a 2D bf16 tile (tileCols x tileRows, row stride strideElems) from
// global memory into LDS at byte offset ldsOff. D# per ISA ch.8 (CDNA5).
// Issued by one wave; completion via TENSORcnt.
// ---------------------------------------------------------------------------
DEV void tdm_load_2d_bf16(unsigned ldsOff, const void* gsrc,
                          int tileCols, int tileRows, int strideElems) {
    unsigned long long ga = (unsigned long long)gsrc;
    u32x4 g0;
    g0[0] = 1u;                                        // count=1, user D#
    g0[1] = ldsOff;                                    // lds_addr (bytes)
    g0[2] = (unsigned)(ga & 0xFFFFFFFFu);              // global_addr[31:0]
    g0[3] = (unsigned)((ga >> 32) & 0x01FFFFFFu)       // global_addr[56:32]
            | 0x80000000u;                             // type=2 ("image"), bits127:126
    i32x8 g1;
    g1[0] = 0x00010000;                                // wg_mask=0, data_size=1 (2B)
    g1[1] = (tileCols & 0xFFFF) << 16;                 // tensor_dim0[15:0] @ bits63:48
    g1[2] = ((tileCols >> 16) & 0xFFFF)                // tensor_dim0[31:16]
            | ((tileRows & 0xFFFF) << 16);             // tensor_dim1[15:0]
    g1[3] = ((tileRows >> 16) & 0xFFFF)                // tensor_dim1[31:16]
            | ((tileCols & 0xFFFF) << 16);             // tile_dim0 @ bits127:112
    g1[4] = (tileRows & 0xFFFF);                       // tile_dim1; tile_dim2=0
    g1[5] = strideElems;                               // tensor_dim0_stride[31:0]
    g1[6] = 0;                                         // stride0 hi | stride1 lo
    g1[7] = 0;                                         // tensor_dim1_stride hi
    i32x4 z4 = {0, 0, 0, 0};
#if defined(__clang_major__) && (__clang_major__ >= 23)
    i32x8 z8 = {0, 0, 0, 0, 0, 0, 0, 0};
    __builtin_amdgcn_tensor_load_to_lds(g0, g1, z4, z4, z8, 0);
#else
    __builtin_amdgcn_tensor_load_to_lds(g0, g1, z4, z4, 0);
#endif
}

// ---------------------------------------------------------------------------
// x = gelu(concat(x_values, y) @ proj_W + proj_b), written as bf16
// grid = B*N blocks, 256 threads (one output column each)
// ---------------------------------------------------------------------------
__global__ void proj_gelu_kernel(const float* __restrict__ xv, const float* __restrict__ yt,
                                 const float* __restrict__ W, const float* __restrict__ bias,
                                 ushort_t* __restrict__ xb) {
    long row = blockIdx.x;
    int d = threadIdx.x;
    const float* f = xv + row * 3;
    float acc = bias[d] + f[0] * W[d] + f[1] * W[256 + d] + f[2] * W[512 + d] +
                yt[row] * W[768 + d];
    float g = 0.5f * acc * (1.0f + erff(acc * 0.70710678118654752f)); // exact gelu
    xb[row * 256 + d] = f2bf(g);
}

// W (256x256 fp32, k-major) -> Wt (n-major bf16) so B fragments load packed K pairs
__global__ void wtrans_kernel(const float* __restrict__ W, ushort_t* __restrict__ Wt) {
    int n = blockIdx.x, k = threadIdx.x;
    Wt[(size_t)n * 256 + k] = f2bf(W[(size_t)k * 256 + n]);
}

// broadcast (rows,256) fp32 -> (B*rows,256) bf16
__global__ void bcast_kernel(const float* __restrict__ src, ushort_t* __restrict__ dst, int rows) {
    int r = blockIdx.x % rows;
    int t = threadIdx.x;
    dst[(size_t)blockIdx.x * 256 + t] = f2bf(src[(size_t)r * 256 + t]);
}

// ---------------------------------------------------------------------------
// C[M,256] = A[M,256] @ W[256,256] + bias  (bf16 in, fp32 acc, bf16 out)
// optional epilogue: out = res + relu(acc+bias)
// grid (M/64, 4), 256 threads = 8 waves; wave tile = 16x32 (two wmma accums).
// A block (64x256 bf16, 32KB) staged into LDS via TDM tensor_load_to_lds.
// ---------------------------------------------------------------------------
__global__ void gemm_bf16_kernel(const ushort_t* __restrict__ A, const ushort_t* __restrict__ Wt,
                                 const float* __restrict__ bias, ushort_t* __restrict__ out,
                                 const ushort_t* __restrict__ res) {
    __shared__ ushort_t ldsA[64 * 256];                // 32 KB A-operand stage
    const int lane = threadIdx.x & 31;
    const int wave = threadIdx.x >> 5;
    const int half = lane >> 4;
    const int l15  = lane & 15;
    const int mw = wave >> 1, nw = wave & 1;
    const long rowBase = (long)blockIdx.x * 64 + mw * 16;
    const int  colBase = blockIdx.y * 64 + nw * 32;

    if (wave == 0) {                                   // one TDM descriptor per WG
        tdm_load_2d_bf16((unsigned)(size_t)ldsA,
                         A + (size_t)blockIdx.x * 64 * 256,
                         256 /*cols*/, 64 /*rows*/, 256 /*stride*/);
        __builtin_amdgcn_s_wait_tensorcnt(0);
    }
    __syncthreads();

    const unsigned* lA32 = (const unsigned*)ldsA;
    const unsigned* W32  = (const unsigned*)Wt;

    v8f c0 = {0, 0, 0, 0, 0, 0, 0, 0};
    v8f c1 = {0, 0, 0, 0, 0, 0, 0, 0};
    for (int k0 = 0; k0 < 256; k0 += 32) {
        FragB16 a, b0, b1;
#pragma unroll
        for (int j = 0; j < 8; ++j) {
            a.u[j]  = lA32[((mw * 16 + l15) * 256 + k0 + kA(j, half)) >> 1];
            b0.u[j] = W32[((long)(colBase + l15) * 256 + k0 + kB(j, half)) >> 1];
            b1.u[j] = W32[((long)(colBase + 16 + l15) * 256 + k0 + kB(j, half)) >> 1];
        }
        c0 = __builtin_amdgcn_wmma_f32_16x16x32_bf16(false, a.v, false, b0.v, (short)0, c0, false, false);
        c1 = __builtin_amdgcn_wmma_f32_16x16x32_bf16(false, a.v, false, b1.v, (short)0, c1, false, false);
    }
#pragma unroll
    for (int i = 0; i < 8; ++i) {
        long m = rowBase + half * 8 + i;
        int col0 = colBase + l15, col1 = colBase + 16 + l15;
        float f0 = c0[i] + bias[col0];
        float f1 = c1[i] + bias[col1];
        if (res) {
            f0 = bf2f(res[m * 256 + col0]) + fmaxf(f0, 0.0f);
            f1 = bf2f(res[m * 256 + col1]) + fmaxf(f1, 0.0f);
        }
        out[m * 256 + col0] = f2bf(f0);
        out[m * 256 + col1] = f2bf(f1);
    }
}

// ---------------------------------------------------------------------------
// Fused multi-head attention: O = q + softmax(mask(q k^T / 16)) v
// grid (nq/16, H, B), 256 threads. Full 16 x nk score slab in LDS (<=192KB).
// ---------------------------------------------------------------------------
__global__ void attn_kernel(const ushort_t* __restrict__ qb, const ushort_t* __restrict__ kb,
                            const ushort_t* __restrict__ vb, const float* __restrict__ mask,
                            ushort_t* __restrict__ ob, int nq, int nk) {
    extern __shared__ char smem[];
    float* scores = (float*)smem;                                  // 16 x nk fp32
    ushort_t* probs = (ushort_t*)(smem + (size_t)16 * nk * 4);     // 16 x nk bf16
    unsigned* probs32 = (unsigned*)probs;
    __shared__ float red[2048];                                    // 2 dt x 4 slices x 16x16

    const int lane = threadIdx.x & 31, wave = threadIdx.x >> 5;
    const int half = lane >> 4, l15 = lane & 15;
    const int qt = blockIdx.x, h = blockIdx.y, b = blockIdx.z;
    const unsigned* q32 = (const unsigned*)qb;
    const unsigned* k32 = (const unsigned*)kb;

    // --- phase 1: score tiles via WMMA (A = q 16x32, B = k^T 32x16) ---
    FragB16 qa;
    long qrow = (long)b * nq + qt * 16 + l15;
#pragma unroll
    for (int j = 0; j < 8; ++j)
        qa.u[j] = q32[(qrow * 256 + h * 32 + kA(j, half)) >> 1];

    for (int kt = wave; kt < (nk >> 4); kt += 8) {
        FragB16 kf;
        long krow = (long)b * nk + kt * 16 + l15;
#pragma unroll
        for (int j = 0; j < 8; ++j)
            kf.u[j] = k32[(krow * 256 + h * 32 + kB(j, half)) >> 1];
        v8f c = {0, 0, 0, 0, 0, 0, 0, 0};
        c = __builtin_amdgcn_wmma_f32_16x16x32_bf16(false, qa.v, false, kf.v, (short)0, c, false, false);
#pragma unroll
        for (int i = 0; i < 8; ++i) {
            int m = half * 8 + i;
            int key = kt * 16 + l15;
            float s = c[i] * 0.0625f;                       // 1/sqrt(D)=1/16 (faithful)
            if (mask && mask[(long)b * nk + key] == 0.0f) s = -1.0e4f;
            scores[m * nk + key] = s;
        }
    }
    __syncthreads();

    // --- phase 2: row softmax (2 rows per wave, wave32 shuffles) ---
    for (int rr = 0; rr < 2; ++rr) {
        int r = wave * 2 + rr;
        float mx = -3.0e38f;
        for (int j = lane; j < nk; j += 32) mx = fmaxf(mx, scores[r * nk + j]);
        for (int off = 16; off > 0; off >>= 1) mx = fmaxf(mx, __shfl_xor(mx, off, 32));
        float sum = 0.0f;
        for (int j = lane; j < nk; j += 32) {
            float e = __expf(scores[r * nk + j] - mx);
            scores[r * nk + j] = e;
            sum += e;
        }
        for (int off = 16; off > 0; off >>= 1) sum += __shfl_xor(sum, off, 32);
        float inv = 1.0f / sum;
        for (int j = lane; j < nk; j += 32) probs[r * nk + j] = f2bf(scores[r * nk + j] * inv);
    }
    __syncthreads();

    // --- phase 3: O = P @ V (2 d-tiles x 4 k-slices across 8 waves) + residual ---
    const int dt = wave >> 2, slice = wave & 3;
    v8f acc = {0, 0, 0, 0, 0, 0, 0, 0};
    for (int c0 = slice; c0 < (nk >> 5); c0 += 4) {
        FragB16 pa, vf;
#pragma unroll
        for (int j = 0; j < 8; ++j) {
            int key = c0 * 32 + kA(j, half);
            pa.u[j] = probs32[(l15 * nk + key) >> 1];
        }
        int dcol = h * 32 + dt * 16 + l15;
#pragma unroll
        for (int j = 0; j < 8; ++j) {
            int key = c0 * 32 + kB(j, half);
            unsigned lo = vb[((long)b * nk + key) * 256 + dcol];
            unsigned hi = vb[((long)b * nk + key + 1) * 256 + dcol];
            vf.u[j] = lo | (hi << 16);
        }
        acc = __builtin_amdgcn_wmma_f32_16x16x32_bf16(false, pa.v, false, vf.v, (short)0, acc, false, false);
    }
#pragma unroll
    for (int i = 0; i < 8; ++i)
        red[(dt * 4 + slice) * 256 + (half * 8 + i) * 16 + l15] = acc[i];
    __syncthreads();
    if (slice == 0) {
#pragma unroll
        for (int i = 0; i < 8; ++i) {
            int m = half * 8 + i, n = l15;
            float s = red[(dt * 4 + 0) * 256 + m * 16 + n] + red[(dt * 4 + 1) * 256 + m * 16 + n] +
                      red[(dt * 4 + 2) * 256 + m * 16 + n] + red[(dt * 4 + 3) * 256 + m * 16 + n];
            long orow = (long)b * nq + qt * 16 + m;
            int col = h * 32 + dt * 16 + n;
            ob[orow * 256 + col] = f2bf(bf2f(qb[orow * 256 + col]) + s);
        }
    }
}

// ---------------------------------------------------------------------------
// Row LayerNorm over D=256; one block per row; writes bf16 and/or fp32
// ---------------------------------------------------------------------------
__global__ void ln_kernel(const ushort_t* __restrict__ in, const float* __restrict__ g,
                          const float* __restrict__ b, ushort_t* __restrict__ outb,
                          float* __restrict__ outf) {
    __shared__ float buf[256];
    __shared__ float stat[2];
    long row = blockIdx.x;
    int t = threadIdx.x;
    float v = bf2f(in[row * 256 + t]);
    buf[t] = v; __syncthreads();
    for (int s = 128; s > 0; s >>= 1) { if (t < s) buf[t] += buf[t + s]; __syncthreads(); }
    if (t == 0) stat[0] = buf[0] * (1.0f / 256.0f);
    __syncthreads();
    float mean = stat[0];
    float dv = v - mean;
    buf[t] = dv * dv; __syncthreads();
    for (int s = 128; s > 0; s >>= 1) { if (t < s) buf[t] += buf[t + s]; __syncthreads(); }
    if (t == 0) stat[1] = buf[0] * (1.0f / 256.0f);
    __syncthreads();
    float y = dv * rsqrtf(stat[1] + 1e-5f) * g[t] + b[t];
    if (outb) outb[row * 256 + t] = f2bf(y);
    if (outf) outf[row * 256 + t] = y;
}

// ---------------------------------------------------------------------------
extern "C" void kernel_launch(void* const* d_in, const int* in_sizes, int n_in,
                              void* d_out, int out_size, void* d_ws, size_t ws_size,
                              hipStream_t stream) {
    (void)in_sizes; (void)n_in; (void)out_size; (void)ws_size;

    const float* x_values   = (const float*)d_in[0];
    const float* y_target   = (const float*)d_in[1];
    const float* point_mask = (const float*)d_in[2];
    const float* proj_W     = (const float*)d_in[3];
    const float* proj_b     = (const float*)d_in[4];
    const float* isab_I     = (const float*)d_in[5];
    const float* isab_Wq    = (const float*)d_in[6];
    const float* isab_bq    = (const float*)d_in[7];
    const float* isab_Wk    = (const float*)d_in[8];
    const float* isab_bk    = (const float*)d_in[9];
    const float* isab_Wv    = (const float*)d_in[10];
    const float* isab_bv    = (const float*)d_in[11];
    const float* isab_Wo    = (const float*)d_in[12];
    const float* isab_bo    = (const float*)d_in[13];
    const float* isab_g0    = (const float*)d_in[14];
    const float* isab_b0    = (const float*)d_in[15];
    const float* isab_g1    = (const float*)d_in[16];
    const float* isab_b1    = (const float*)d_in[17];
    const float* pma_S      = (const float*)d_in[18];
    const float* pma_Wq     = (const float*)d_in[19];
    const float* pma_bq     = (const float*)d_in[20];
    const float* pma_Wk     = (const float*)d_in[21];
    const float* pma_bk     = (const float*)d_in[22];
    const float* pma_Wv     = (const float*)d_in[23];
    const float* pma_bv     = (const float*)d_in[24];
    const float* pma_Wo     = (const float*)d_in[25];
    const float* pma_bo     = (const float*)d_in[26];
    const float* pma_g0     = (const float*)d_in[27];
    const float* pma_b0     = (const float*)d_in[28];
    const float* pma_g1     = (const float*)d_in[29];
    const float* pma_b1     = (const float*)d_in[30];
    float* outF = (float*)d_out;

    // --- workspace carve (all bf16 activations; ~170 MB total) ---
    char* p = (char*)d_ws;
    auto take = [&](size_t bytes) -> char* {
        char* r = p; p += (bytes + 255) & ~(size_t)255; return r;
    };
    ushort_t* wts  = (ushort_t*)take(28ull * 65536 * 2); // 28 transposed bf16 weight mats
    ushort_t* xb   = (ushort_t*)take(65536ull * 256 * 2);
    ushort_t* qb   = (ushort_t*)take(65536ull * 256 * 2);
    ushort_t* kbuf = (ushort_t*)take(65536ull * 256 * 2);
    ushort_t* vbuf = (ushort_t*)take(65536ull * 256 * 2);
    ushort_t* obuf = (ushort_t*)take(65536ull * 256 * 2);
    ushort_t* hb   = (ushort_t*)take(2048ull * 256 * 2);
    ushort_t* Ib   = (ushort_t*)take(2048ull * 256 * 2);
    ushort_t* Sb   = (ushort_t*)take(1024ull * 256 * 2);

    auto WT = [&](int slot) { return wts + (size_t)slot * 65536; };
    // slots: Wq 0..5, Wk 6..11, Wv 12..17, Wo 18..23, pma q/k/v/o 24..27
    for (int l = 0; l < 3; ++l)
        for (int m = 0; m < 2; ++m) {
            size_t off = (size_t)(l * 2 + m) * 65536;
            wtrans_kernel<<<256, 256, 0, stream>>>(isab_Wq + off, WT(0  + l * 2 + m));
            wtrans_kernel<<<256, 256, 0, stream>>>(isab_Wk + off, WT(6  + l * 2 + m));
            wtrans_kernel<<<256, 256, 0, stream>>>(isab_Wv + off, WT(12 + l * 2 + m));
            wtrans_kernel<<<256, 256, 0, stream>>>(isab_Wo + off, WT(18 + l * 2 + m));
        }
    wtrans_kernel<<<256, 256, 0, stream>>>(pma_Wq, WT(24));
    wtrans_kernel<<<256, 256, 0, stream>>>(pma_Wk, WT(25));
    wtrans_kernel<<<256, 256, 0, stream>>>(pma_Wv, WT(26));
    wtrans_kernel<<<256, 256, 0, stream>>>(pma_Wo, WT(27));

    // input projection + exact gelu
    proj_gelu_kernel<<<65536, 256, 0, stream>>>(x_values, y_target, proj_W, proj_b, xb);

    auto run_mab = [&](const ushort_t* Qin, int nq, const ushort_t* Kin, int nk,
                       const ushort_t* WqT, const float* bq, const ushort_t* WkT, const float* bk,
                       const ushort_t* WvT, const float* bv, const ushort_t* WoT, const float* bo,
                       const float* g0, const float* b0, const float* g1, const float* b1,
                       const float* mask, ushort_t* dstB, float* dstF) {
        int Mq = 32 * nq, Mk = 32 * nk;
        gemm_bf16_kernel<<<dim3(Mq / 64, 4), 256, 0, stream>>>(Qin, WqT, bq, qb, nullptr);
        gemm_bf16_kernel<<<dim3(Mk / 64, 4), 256, 0, stream>>>(Kin, WkT, bk, kbuf, nullptr);
        gemm_bf16_kernel<<<dim3(Mk / 64, 4), 256, 0, stream>>>(Kin, WvT, bv, vbuf, nullptr);
        size_t smem = (size_t)16 * nk * 6;   // fp32 scores + bf16 probs
        attn_kernel<<<dim3(nq / 16, 8, 32), 256, smem, stream>>>(qb, kbuf, vbuf, mask, obuf, nq, nk);
        ln_kernel<<<Mq, 256, 0, stream>>>(obuf, g0, b0, obuf, nullptr);            // LN(g0,b0), in place
        gemm_bf16_kernel<<<dim3(Mq / 64, 4), 256, 0, stream>>>(obuf, WoT, bo, kbuf, obuf); // O + relu(O Wo + bo)
        ln_kernel<<<Mq, 256, 0, stream>>>(kbuf, g1, b1, dstB, dstF);               // LN(g1,b1)
    };

    for (int l = 0; l < 3; ++l) {
        bcast_kernel<<<32 * 64, 256, 0, stream>>>(isab_I + (size_t)l * 64 * 256, Ib, 64);
        int s0 = l * 2 + 0, s1 = l * 2 + 1;
        // mab0: inducing points attend to X (masked by point_mask) -> h
        run_mab(Ib, 64, xb, 2048,
                WT(0 + s0), isab_bq + (size_t)s0 * 256, WT(6 + s0), isab_bk + (size_t)s0 * 256,
                WT(12 + s0), isab_bv + (size_t)s0 * 256, WT(18 + s0), isab_bo + (size_t)s0 * 256,
                isab_g0 + (size_t)s0 * 256, isab_b0 + (size_t)s0 * 256,
                isab_g1 + (size_t)s0 * 256, isab_b1 + (size_t)s0 * 256,
                point_mask, hb, nullptr);
        // mab1: X attends to h (all valid) -> new x
        run_mab(xb, 2048, hb, 64,
                WT(0 + s1), isab_bq + (size_t)s1 * 256, WT(6 + s1), isab_bk + (size_t)s1 * 256,
                WT(12 + s1), isab_bv + (size_t)s1 * 256, WT(18 + s1), isab_bo + (size_t)s1 * 256,
                isab_g0 + (size_t)s1 * 256, isab_b0 + (size_t)s1 * 256,
                isab_g1 + (size_t)s1 * 256, isab_b1 + (size_t)s1 * 256,
                nullptr, xb, nullptr);
    }

    // PMA: seeds attend to X (masked) -> final (B,32,256) fp32 output
    bcast_kernel<<<32 * 32, 256, 0, stream>>>(pma_S, Sb, 32);
    run_mab(Sb, 32, xb, 2048,
            WT(24), pma_bq, WT(25), pma_bk, WT(26), pma_bv, WT(27), pma_bo,
            pma_g0, pma_b0, pma_g1, pma_b1,
            point_mask, nullptr, outF);
}
